// MultiHeadAttention_73443940761660
// MI455X (gfx1250) — matrix operand order repaired and checked by reading, not compiled, not run on hardware
//
#include <hip/hip_runtime.h>
#include <hip/hip_bf16.h>

typedef __attribute__((ext_vector_type(16))) _Float16 v16h;
typedef __attribute__((ext_vector_type(8)))  _Float16 v8h;
typedef __attribute__((ext_vector_type(8)))  float    v8f;

#define N_HEADS   16
#define HEAD_SIZE 64
#define SEQ_LEN   2048
#define EMBED     1024
#define BATCH     2
#define D_INNER   1024                 // N_HEADS * HEAD_SIZE
#define M_ROWS    (BATCH * SEQ_LEN)    // 4096
#define LOG2E     1.44269504088896340736f

// ---------------------------------------------------------------------------
// A-fragment (16x32 f16, MxK): lane<16 -> row=lane, elements {kb..kb+15}
// with kb = (lane>=16)?16:0.  Two contiguous 16B chunks starting at p (=base+kb).
// ---------------------------------------------------------------------------
__device__ inline v16h load_afrag16(const _Float16* p) {
    v16h a;
    *reinterpret_cast<v8h*>(&a)       = *reinterpret_cast<const v8h*>(p);
    *(reinterpret_cast<v8h*>(&a) + 1) = *reinterpret_cast<const v8h*>(p + 8);
    return a;
}

__device__ inline v8f wmma_f16(v16h a, v16h b, v8f c) {
    return __builtin_amdgcn_wmma_f32_16x16x32_f16(false, a, false, b,
                                                  (short)0, c, false, false);
}

// ---------------------------------------------------------------------------
// f32 -> f16 elementwise convert (4 elems/thread)
// ---------------------------------------------------------------------------
__global__ void cvt_f32_to_f16(const float* __restrict__ in,
                               _Float16* __restrict__ out, int n) {
    int i = (blockIdx.x * blockDim.x + threadIdx.x) * 4;
    if (i + 3 < n) {
        float4 f = *(const float4*)(in + i);
        out[i + 0] = (_Float16)f.x;
        out[i + 1] = (_Float16)f.y;
        out[i + 2] = (_Float16)f.z;
        out[i + 3] = (_Float16)f.w;
    }
}

// ---------------------------------------------------------------------------
// Weight convert + transpose: out[n][k] = (f16) in[k][n], 1024x1024
// ---------------------------------------------------------------------------
__global__ void transpose_cvt(const float* __restrict__ in,
                              _Float16* __restrict__ out) {
    __shared__ float tile[16][17];
    int k = blockIdx.y * 16 + threadIdx.y;
    int n = blockIdx.x * 16 + threadIdx.x;
    tile[threadIdx.y][threadIdx.x] = in[(size_t)k * D_INNER + n];
    __syncthreads();
    int on = blockIdx.x * 16 + threadIdx.y;
    int ok = blockIdx.y * 16 + threadIdx.x;
    out[(size_t)on * EMBED + ok] = (_Float16)tile[threadIdx.x][threadIdx.y];
}

// ---------------------------------------------------------------------------
// GEMM: C = A[M_ROWS x 1024](f16) * B (as BT[n][k], f16); out = (C+bias)*scale
// Per wave: 64x32 tile (4x2 WMMA tiles).  WG = 8 waves -> 256x64.  Grid (16,16).
// MODE 0: f16 row-major.  MODE 1: f16 transposed-V [B,H,Dh,S].  MODE 2: f32.
// ---------------------------------------------------------------------------
template<int MODE>
__global__ __launch_bounds__(256) void gemm_wmma(
        const _Float16* __restrict__ A, const _Float16* __restrict__ BT,
        const float* __restrict__ bias, void* __restrict__ outp, float scale) {
    const int lane = threadIdx.x & 31;
    const int w    = threadIdx.x >> 5;
    const int wm   = w & 3, wn = w >> 2;
    const int row0 = blockIdx.x * 256 + wm * 64;
    const int col0 = blockIdx.y * 64 + wn * 32;
    const int l15  = lane & 15;
    const int hi   = lane >> 4;
    const int kbA  = hi * 16;
    const int kbB  = hi * 16;
    const int mb   = hi * 8;

    v8f acc[4][2];
    #pragma unroll
    for (int mt = 0; mt < 4; ++mt)
        #pragma unroll
        for (int nt = 0; nt < 2; ++nt)
            #pragma unroll
            for (int v = 0; v < 8; ++v) acc[mt][nt][v] = 0.f;

    for (int k0 = 0; k0 < EMBED; k0 += 32) {
        v16h a[4], b[2];
        #pragma unroll
        for (int mt = 0; mt < 4; ++mt)
            a[mt] = load_afrag16(A + (size_t)(row0 + mt * 16 + l15) * EMBED + k0 + kbA);
        #pragma unroll
        for (int nt = 0; nt < 2; ++nt)
            b[nt] = *reinterpret_cast<const v16h*>(
                        BT + (size_t)(col0 + nt * 16 + l15) * EMBED + k0 + kbB);
        #pragma unroll
        for (int mt = 0; mt < 4; ++mt)
            #pragma unroll
            for (int nt = 0; nt < 2; ++nt)
                acc[mt][nt] = wmma_f16(a[mt], b[nt], acc[mt][nt]);
    }

    #pragma unroll
    for (int nt = 0; nt < 2; ++nt) {
        const int col = col0 + nt * 16 + l15;
        const float bv = bias[col];
        #pragma unroll
        for (int mt = 0; mt < 4; ++mt) {
            #pragma unroll
            for (int v = 0; v < 8; ++v) {
                const int row = row0 + mt * 16 + mb + v;
                const float val = (acc[mt][nt][v] + bv) * scale;
                if (MODE == 0) {
                    ((_Float16*)outp)[(size_t)row * D_INNER + col] = (_Float16)val;
                } else if (MODE == 1) {
                    const int bat = row >> 11, s = row & (SEQ_LEN - 1);
                    ((_Float16*)outp)[((size_t)bat * D_INNER + col) * SEQ_LEN + s] =
                        (_Float16)val;
                } else {
                    ((float*)outp)[(size_t)row * EMBED + col] = val;
                }
            }
        }
    }
}

// ---------------------------------------------------------------------------
// Flash attention, transposed-score orientation, 64-key steps.
// S^T = K * Q^T  (lane = query column, per-lane regs = keys), so softmax
// row-stats are in-lane reductions + one shfl_xor(16).  Q is pre-scaled by
// log2(e)/8, so scores are in log2 units and p = v_exp(s - m) directly.
// P^T converts to PV B-fragments with one 16B lane-half exchange per 32 keys.
// out^T = V^T * P^T.   Q,K: [B,S,H*Dh] f16.  Vt: [B,H,Dh,S] f16.
// ---------------------------------------------------------------------------
template<bool MASK>
__device__ inline void fa_step(int j, const _Float16* __restrict__ Kp,
                               const _Float16* __restrict__ Vp,
                               const v16h (&qb)[2], v8f (&acc)[4],
                               float& m, float& lsum,
                               int l15, int hi, int q0) {
    const int kbA = hi * 16;

    // ---- S^T tiles: keys [j + t*16, j + t*16 + 16) x queries [q0, q0+16) ----
    v8f s[4];
    #pragma unroll
    for (int t = 0; t < 4; ++t) {
        #pragma unroll
        for (int v = 0; v < 8; ++v) s[t][v] = 0.f;
        #pragma unroll
        for (int hh = 0; hh < 2; ++hh) {
            v16h ka = load_afrag16(
                Kp + (size_t)(j + t * 16 + l15) * D_INNER + hh * 32 + kbA);
            s[t] = wmma_f16(ka, qb[hh], s[t]);
        }
    }

    // ---- mask (diagonal step only) ----
    #pragma unroll
    for (int t = 0; t < 4; ++t)
        #pragma unroll
        for (int v = 0; v < 8; ++v) {
            if (MASK) {
                const int key = j + t * 16 + hi * 8 + v;
                if (key > q0 + l15) s[t][v] = -1e30f;
            }
        }

    // ---- online softmax: in-lane reduce over 32 keys + 1 lane-half merge ----
    float rm = s[0][0];
    #pragma unroll
    for (int t = 0; t < 4; ++t)
        #pragma unroll
        for (int v = 0; v < 8; ++v) rm = fmaxf(rm, s[t][v]);
    rm = fmaxf(rm, __shfl_xor(rm, 16, 32));
    const float mn    = fmaxf(m, rm);
    const float alpha = __builtin_amdgcn_exp2f(m - mn);   // log2-units state

    v8h ph[4];
    float rs = 0.f;
    #pragma unroll
    for (int t = 0; t < 4; ++t)
        #pragma unroll
        for (int v = 0; v < 8; ++v) {
            const float p = __builtin_amdgcn_exp2f(s[t][v] - mn);
            rs += p;
            ph[t][v] = (_Float16)p;
        }
    rs += __shfl_xor(rs, 16, 32);
    lsum = lsum * alpha + rs;
    m = mn;
    #pragma unroll
    for (int n = 0; n < 4; ++n)
        #pragma unroll
        for (int v = 0; v < 8; ++v) acc[n][v] *= alpha;

    // ---- P^T -> B-fragments (one per 32-key group): lane-half exchange ----
    v16h pb[2];
    #pragma unroll
    for (int u = 0; u < 2; ++u) {
        v8h keep = hi ? ph[2 * u + 1] : ph[2 * u];   // keys u*32 + hi*16 + hi*8 + v
        v8h send = hi ? ph[2 * u]     : ph[2 * u + 1];
        int si[4], ri[4];
        __builtin_memcpy(si, &send, 16);
        #pragma unroll
        for (int d = 0; d < 4; ++d) ri[d] = __shfl_xor(si[d], 16, 32);
        v8h recv;
        __builtin_memcpy(&recv, ri, 16);
        *reinterpret_cast<v8h*>(&pb[u])       = hi ? recv : keep;  // keys +0..7
        *(reinterpret_cast<v8h*>(&pb[u]) + 1) = hi ? keep : recv;  // keys +8..15
    }

    // ---- out^T += V^T_tile * P^T : 4 d-chunks x 2 key-groups ----
    #pragma unroll
    for (int n = 0; n < 4; ++n)
        #pragma unroll
        for (int u = 0; u < 2; ++u) {
            v16h va = load_afrag16(
                Vp + (size_t)(n * 16 + l15) * SEQ_LEN + j + u * 32 + kbA);
            acc[n] = wmma_f16(va, pb[u], acc[n]);
        }
}

__global__ __launch_bounds__(256) void flash_attn(
        const _Float16* __restrict__ Q, const _Float16* __restrict__ K,
        const _Float16* __restrict__ Vt, _Float16* __restrict__ O) {
    const int lane = threadIdx.x & 31;
    const int w    = threadIdx.x >> 5;
    const int l15  = lane & 15;
    const int hi   = lane >> 4;
    const int qblk = blockIdx.x & 15;          // SEQ_LEN/128 = 16
    const int h    = (blockIdx.x >> 4) & 15;
    const int bb   = blockIdx.x >> 8;
    const int q0   = qblk * 128 + w * 16;

    const _Float16* Qp = Q + (size_t)bb * SEQ_LEN * D_INNER + h * HEAD_SIZE;
    const _Float16* Kp = K + (size_t)bb * SEQ_LEN * D_INNER + h * HEAD_SIZE;
    const _Float16* Vp = Vt + (size_t)(bb * N_HEADS + h) * HEAD_SIZE * SEQ_LEN;

    const int kbB = hi * 16;

    // Q^T B-fragments (col = query, elements = d), loaded once
    v16h qb[2];
    #pragma unroll
    for (int hh = 0; hh < 2; ++hh)
        qb[hh] = *reinterpret_cast<const v16h*>(
            Qp + (size_t)(q0 + l15) * D_INNER + hh * 32 + kbB);

    v8f acc[4];
    #pragma unroll
    for (int n = 0; n < 4; ++n)
        #pragma unroll
        for (int v = 0; v < 8; ++v) acc[n][v] = 0.f;
    float m = -3.0e38f, lsum = 0.f;

    const int jm = q0 & ~63;                   // fully-unmasked range
    for (int j = 0; j < jm; j += 64)
        fa_step<false>(j, Kp, Vp, qb, acc, m, lsum, l15, hi, q0);
    for (int j = jm; j < q0 + 16; j += 64)     // exactly 1 diagonal step
        fa_step<true>(j, Kp, Vp, qb, acc, m, lsum, l15, hi, q0);

    // ---- normalize + store: 8 contiguous f16 per lane per d-chunk ----
    const float inv = 1.f / lsum;
    #pragma unroll
    for (int n = 0; n < 4; ++n) {
        v8h o;
        #pragma unroll
        for (int v = 0; v < 8; ++v) o[v] = (_Float16)(acc[n][v] * inv);
        *reinterpret_cast<v8h*>(
            O + (size_t)(bb * SEQ_LEN + q0 + l15) * D_INNER +
            h * HEAD_SIZE + n * 16 + hi * 8) = o;
    }
}

// ---------------------------------------------------------------------------
extern "C" void kernel_launch(void* const* d_in, const int* in_sizes, int n_in,
                              void* d_out, int out_size, void* d_ws, size_t ws_size,
                              hipStream_t stream) {
    const float* x  = (const float*)d_in[0];
    const float* Wq = (const float*)d_in[1];
    const float* bq = (const float*)d_in[2];
    const float* Wk = (const float*)d_in[3];
    const float* bk = (const float*)d_in[4];
    const float* Wv = (const float*)d_in[5];
    const float* bv = (const float*)d_in[6];
    const float* Wo = (const float*)d_in[7];
    const float* bo = (const float*)d_in[8];

    _Float16* hws = (_Float16*)d_ws;
    const size_t MN = (size_t)M_ROWS * EMBED;      // 4M halves
    const size_t WN = (size_t)EMBED * D_INNER;     // 1M halves
    _Float16* Xh   = hws;             // activations f16; later reused as attn out
    _Float16* WqT  = hws + MN;
    _Float16* WkT  = WqT + WN;
    _Float16* WvT  = WkT + WN;
    _Float16* WoT  = WvT + WN;
    _Float16* Qh   = WoT + WN;
    _Float16* Kh   = Qh + MN;
    _Float16* Vth  = Kh + MN;         // total: 40 MB

    const int nX = M_ROWS * EMBED;
    cvt_f32_to_f16<<<nX / (256 * 4), 256, 0, stream>>>(x, Xh, nX);

    dim3 tb(16, 16), tg(EMBED / 16, EMBED / 16);
    transpose_cvt<<<tg, tb, 0, stream>>>(Wq, WqT);
    transpose_cvt<<<tg, tb, 0, stream>>>(Wk, WkT);
    transpose_cvt<<<tg, tb, 0, stream>>>(Wv, WvT);
    transpose_cvt<<<tg, tb, 0, stream>>>(Wo, WoT);

    dim3 gg(M_ROWS / 256, D_INNER / 64);           // (16,16)
    // Q pre-scaled by log2(e)/sqrt(64): scores come out in log2 units.
    gemm_wmma<0><<<gg, 256, 0, stream>>>(Xh, WqT, bq, Qh, 0.125f * LOG2E);
    gemm_wmma<0><<<gg, 256, 0, stream>>>(Xh, WkT, bk, Kh, 1.0f);
    gemm_wmma<1><<<gg, 256, 0, stream>>>(Xh, WvT, bv, Vth, 1.0f);

    flash_attn<<<BATCH * N_HEADS * (SEQ_LEN / 128), 256, 0, stream>>>(Qh, Kh, Vth, Xh);

    gemm_wmma<2><<<gg, 256, 0, stream>>>(Xh, WoT, bo, d_out, 1.0f);
}
